// SelfAttention_5050881540600
// MI455X (gfx1250) — matrix operand order repaired and checked by reading, not compiled, not run on hardware
//
#include <hip/hip_runtime.h>
#include <hip/hip_bf16.h>

// ---------------------------------------------------------------------------
// Fused self-attention for MI455X (gfx1250, wave32, WMMA + async-to-LDS DMA).
//   B=8, H=W=64 -> N=4096, C=256, Cf=32
// Pipeline:
//   K1: pack Wf|Wg|Wh -> bf16 WcatT[320][256] (transposed: row=out-ch, col=in-ch)
//   K2: WMMA GEMM  fgh = x @ Wcat   (M=32768, K=256, Nout=320)
//       f,g -> fg[B*N][64] row-major (bf16);  h -> hT[B][256][4096] (bf16, transposed)
//   K3: cooperative flash-attention: 8-wave block = 4 query tiles x 2 col halves,
//       32-key blocks streamed through a double-buffered LDS h-tile filled with
//       global_load_async_to_lds_b128 (ASYNCcnt), S/P/O all via wmma_f32_16x16x32_bf16.
// ---------------------------------------------------------------------------

typedef __attribute__((ext_vector_type(16))) __bf16 v16bf;
typedef __attribute__((ext_vector_type(8)))  float  v8f;

union Frag16 { uint4 u[2]; v16bf v; };   // 16 bf16 = two 16B loads
union Pack8  { uint4 u; __bf16 h[8]; };  // 8 bf16 packed store

#define WMMA_BF16(a, b, c) \
  __builtin_amdgcn_wmma_f32_16x16x32_bf16(false, (a), false, (b), (short)0, (c), false, false)

static constexpr int kB  = 8;
static constexpr int kN  = 4096;   // H*W
static constexpr int kC  = 256;
static constexpr int kNT = 320;    // 32 (f) + 32 (g) + 256 (h)

// ---------------------------------------------------------------------------
// Kernel 1: weights f32 -> transposed concatenated bf16  WcatT[d][c], d in [0,320)
// ---------------------------------------------------------------------------
__global__ void wconv_kernel(const float* __restrict__ kf, const float* __restrict__ kg,
                             const float* __restrict__ kh, __bf16* __restrict__ W) {
  int idx = blockIdx.x * blockDim.x + threadIdx.x;
  if (idx >= kNT * kC) return;
  int d = idx >> 8;          // out channel 0..319
  int c = idx & 255;         // in channel
  float w;
  if (d < 32)      w = kf[c * 32 + d];
  else if (d < 64) w = kg[c * 32 + (d - 32)];
  else             w = kh[c * 256 + (d - 64)];
  W[(long)d * 256 + c] = (__bf16)w;
}

// ---------------------------------------------------------------------------
// Kernel 2: projection GEMM via WMMA bf16.
// One wave computes a 16-row x 64-col output slab (4 accumulator tiles),
// looping K=256 in 8 steps of 32. x is converted f32->bf16 in registers.
// Grid: 2048 row-tiles * 5 col-groups = 10240 waves = 1280 blocks of 256.
// ---------------------------------------------------------------------------
__global__ __launch_bounds__(256) void proj_kernel(const float* __restrict__ x,
                                                   const __bf16* __restrict__ W,
                                                   __bf16* __restrict__ fg,
                                                   __bf16* __restrict__ hT) {
  int w    = blockIdx.x * (blockDim.x >> 5) + (threadIdx.x >> 5);
  int lane = threadIdx.x & 31;
  int ln   = lane & 15;
  int hi   = lane >> 4;          // 0: lanes 0-15, 1: lanes 16-31
  int rowTile = w / 5;
  int cg      = w % 5;           // column group: 0 -> f|g, 1..4 -> h
  long rowBase = (long)rowTile * 16;

  v8f acc[4] = {};
  const float* xrow = x + (rowBase + ln) * kC;
  int b0 = hi * 8;               // A-matrix K base for this half-wave

  for (int ks = 0; ks < 8; ++ks) {
    // A fragment: row m=ln, K runs [b0..b0+7] and [16+b0..16+b0+7] of this K-step.
    const float* xp = xrow + ks * 32 + b0;
    float4 a0 = *(const float4*)(xp);
    float4 a1 = *(const float4*)(xp + 4);
    float4 a2 = *(const float4*)(xp + 16);
    float4 a3 = *(const float4*)(xp + 20);
    v16bf af;
    af[0]  = (__bf16)a0.x; af[1]  = (__bf16)a0.y; af[2]  = (__bf16)a0.z; af[3]  = (__bf16)a0.w;
    af[4]  = (__bf16)a1.x; af[5]  = (__bf16)a1.y; af[6]  = (__bf16)a1.z; af[7]  = (__bf16)a1.w;
    af[8]  = (__bf16)a2.x; af[9]  = (__bf16)a2.y; af[10] = (__bf16)a2.z; af[11] = (__bf16)a2.w;
    af[12] = (__bf16)a3.x; af[13] = (__bf16)a3.y; af[14] = (__bf16)a3.z; af[15] = (__bf16)a3.w;
#pragma unroll
    for (int j = 0; j < 4; ++j) {
      // B fragment: col n=ln, K runs [hi*8..hi*8+7] and [16+hi*8..16+hi*8+7] along WcatT row.
      int d0 = cg * 64 + j * 16 + ln;
      const __bf16* wp = W + (long)d0 * 256 + ks * 32 + hi * 8;
      Frag16 bf;
      bf.u[0] = *(const uint4*)(wp);
      bf.u[1] = *(const uint4*)(wp + 16);
      acc[j] = WMMA_BF16(af, bf.v, acc[j]);
    }
  }

  if (cg == 0) {
    // f and g: store row-major [n][64]  (cols 0..31 = f, 32..63 = g)
#pragma unroll
    for (int j = 0; j < 4; ++j)
#pragma unroll
      for (int v = 0; v < 8; ++v) {
        long r = rowBase + v + hi * 8;                 // C-layout: row v (+8 hi half)
        fg[r * 64 + j * 16 + ln] = (__bf16)acc[j][v];
      }
  } else {
    // h: store transposed hT[b][d][n]; per lane the 8 rows are consecutive n
    // -> one packed 16B store per tile (free transpose from C-layout).
    int bIdx = (int)(rowBase >> 12);
    int nInB = (int)(rowBase & 4095) + hi * 8;
#pragma unroll
    for (int j = 0; j < 4; ++j) {
      int d = cg * 64 + j * 16 + ln - 64;
      Pack8 p;
#pragma unroll
      for (int v = 0; v < 8; ++v) p.h[v] = (__bf16)acc[j][v];
      *(uint4*)(hT + ((long)(bIdx * 256 + d) << 12) + nInB) = p.u;
    }
  }
}

// ---------------------------------------------------------------------------
// Kernel 3: cooperative fused flash attention.
// Block = 8 waves = 4 query tiles x 2 col-halves, one batch. For each 32-key
// block the 16KB h tile is DMA'd into LDS (double-buffered, async-to-LDS,
// overlapped with compute); each wave: 2 WMMAs (S), online softmax, P reshaped
// through LDS, 8 WMMAs (O) against ds_load_b128 h fragments.
// Grid: 8 batches * 64 query groups = 512 blocks of 256.
// ---------------------------------------------------------------------------
__global__ __launch_bounds__(256) void attn_kernel(const float* __restrict__ x,
                                                   const __bf16* __restrict__ fg,
                                                   const __bf16* __restrict__ hT,
                                                   const float* __restrict__ gamma,
                                                   float* __restrict__ out) {
  // h staging: [buf][channel d][32 keys + pad] ; 80B rows -> conflict-free frags
  __shared__ __bf16 hbuf[2][256][40];   // 40,960 B
  __shared__ __bf16 plds[8][16][40];    // 10,240 B (per-wave P tiles)

  int wid  = threadIdx.x >> 5;
  int lane = threadIdx.x & 31;
  int ln   = lane & 15;
  int hi   = lane >> 4;
  int b      = blockIdx.x >> 6;          // batch
  int qGroup = blockIdx.x & 63;          // 4 query tiles per block
  int qsub   = wid >> 1;
  int ch     = wid & 1;                  // 128-col half of C
  int qBase  = (qGroup * 4 + qsub) * 16;
  long rowB  = (long)b * kN;

  const int tix = threadIdx.x;
  // Async DMA of one 32-key h tile (256 ch x 64B) into hbuf[buf]:
  // 1024 16B chunks over 256 lanes -> 4 global_load_async_to_lds_b128 per lane.
  auto issue_h = [&](int buf, int keyBase) {
#pragma unroll
    for (int i = 0; i < 4; ++i) {
      int c    = tix + 256 * i;
      int d    = c >> 2;                 // channel 0..255
      int part = c & 3;                  // 16B chunk within the 64B key run
      const __bf16* gp = hT + ((long)(b * 256 + d) << 12) + keyBase + part * 8;
      unsigned lds = (unsigned)(size_t)&hbuf[buf][d][part * 8];
      asm volatile("global_load_async_to_lds_b128 %0, %1, off"
                   :: "v"(lds), "v"((unsigned long long)(size_t)gp)
                   : "memory");
    }
  };

  // g A-fragment (queries): K = g-dims 0..31 -> fg cols 32..63, two 16B runs.
  Frag16 gf;
  {
    const __bf16* gp = fg + (rowB + qBase + ln) * 64 + 32 + hi * 8;
    gf.u[0] = *(const uint4*)(gp);
    gf.u[1] = *(const uint4*)(gp + 16);
  }

  float mrow[8], lrow[8];
#pragma unroll
  for (int v = 0; v < 8; ++v) { mrow[v] = -__builtin_inff(); lrow[v] = 0.f; }
  v8f acc[8] = {};

  issue_h(0, 0);                         // prime buffer 0

  for (int kb = 0; kb < kN / 32; ++kb) {
    int cur = kb & 1;
    int keyBase = kb * 32;

    // f B-fragments (global, tiny): start them before the async wait.
    const __bf16* fp0 = fg + (rowB + keyBase + ln) * 64 + hi * 8;
    const __bf16* fp1 = fp0 + 16 * 64;
    Frag16 f0, f1;
    f0.u[0] = *(const uint4*)(fp0);     f0.u[1] = *(const uint4*)(fp0 + 16);
    f1.u[0] = *(const uint4*)(fp1);     f1.u[1] = *(const uint4*)(fp1 + 16);

    // Own DMA done -> barrier: everyone's DMA done & prev buffer fully read.
    asm volatile("s_wait_asynccnt 0x0" ::: "memory");
    __syncthreads();
    // Kick off next key block's DMA; it overlaps all compute below.
    if (kb + 1 < kN / 32) issue_h(1 - cur, keyBase + 32);

    v8f z = {};
    v8f S0 = WMMA_BF16(gf.v, f0.v, z);
    v8f S1 = WMMA_BF16(gf.v, f1.v, z);

    // Online softmax over this 32-key block (row stats per C-layout row).
    float sc[8];
#pragma unroll
    for (int v = 0; v < 8; ++v) {
      float mx = fmaxf(S0[v], S1[v]);
      for (int m = 1; m < 16; m <<= 1) mx = fmaxf(mx, __shfl_xor(mx, m, 32));
      float mn = fmaxf(mrow[v], mx);
      float s  = __expf(mrow[v] - mn);
      float p0 = __expf(S0[v] - mn);
      float p1 = __expf(S1[v] - mn);
      float rs = p0 + p1;
      for (int m = 1; m < 16; m <<= 1) rs += __shfl_xor(rs, m, 32);
      lrow[v] = lrow[v] * s + rs;
      mrow[v] = mn;
      sc[v] = s;
      S0[v] = p0; S1[v] = p1;
    }
#pragma unroll
    for (int t = 0; t < 8; ++t)
#pragma unroll
      for (int v = 0; v < 8; ++v) acc[t][v] *= sc[v];

    // Reshape P (C-layout) -> A-layout through per-wave LDS tile.
#pragma unroll
    for (int v = 0; v < 8; ++v) {
      plds[wid][v + hi * 8][ln]      = (__bf16)S0[v];
      plds[wid][v + hi * 8][ln + 16] = (__bf16)S1[v];
    }
    Frag16 pf;
    pf.u[0] = *(const uint4*)&plds[wid][ln][hi * 8];
    pf.u[1] = *(const uint4*)&plds[wid][ln][hi * 8 + 16];

    // Pre-load all 8 h B-fragments from LDS into distinct registers, then run
    // the 8 WMMAs back-to-back (no per-WMMA load wait, no D->A/B hazard).
    Frag16 hf[8];
#pragma unroll
    for (int t = 0; t < 8; ++t) {
      const __bf16* hp = &hbuf[cur][ch * 128 + t * 16 + ln][hi * 8];
      hf[t].u[0] = *(const uint4*)(hp);
      hf[t].u[1] = *(const uint4*)(hp + 16);
    }
#pragma unroll
    for (int t = 0; t < 8; ++t)
      acc[t] = WMMA_BF16(pf.v, hf[t].v, acc[t]);

    if (kb + 1 < kN / 32)   // gfx1250 global_prefetch_b8 of next f rows
      __builtin_prefetch(fg + (rowB + keyBase + 32 + ln) * 64, 0, 1);
  }

  // Epilogue: out = gamma * (O / l) + x
  float gam = gamma[0];
#pragma unroll
  for (int t = 0; t < 8; ++t)
#pragma unroll
    for (int v = 0; v < 8; ++v) {
      long n = (long)b * kN + qBase + v + hi * 8;
      int  c = ch * 128 + t * 16 + ln;
      long idx = n * kC + c;
      out[idx] = gam * (acc[t][v] / lrow[v]) + x[idx];
    }
}

// ---------------------------------------------------------------------------
extern "C" void kernel_launch(void* const* d_in, const int* in_sizes, int n_in,
                              void* d_out, int out_size, void* d_ws, size_t ws_size,
                              hipStream_t stream) {
  const float* x     = (const float*)d_in[0];
  const float* kf    = (const float*)d_in[1];
  const float* kg    = (const float*)d_in[2];
  const float* kh    = (const float*)d_in[3];
  const float* gamma = (const float*)d_in[4];
  float* out = (float*)d_out;

  // Workspace layout (bf16):
  //   WcatT : 320*256           = 163,840 B
  //   fg    : 32768*64          = 4,194,304 B
  //   hT    : 8*256*4096        = 16,777,216 B     (total ~21.1 MB)
  char* ws = (char*)d_ws;
  __bf16* W  = (__bf16*)(ws);
  __bf16* fg = (__bf16*)(ws + 163840);
  __bf16* hT = (__bf16*)(ws + 163840 + 4194304);

  hipLaunchKernelGGL(wconv_kernel, dim3(320),  dim3(256), 0, stream, kf, kg, kh, W);
  hipLaunchKernelGGL(proj_kernel,  dim3(1280), dim3(256), 0, stream, x, W, fg, hT);
  hipLaunchKernelGGL(attn_kernel,  dim3(512),  dim3(256), 0, stream, x, fg, hT, gamma, out);
}
